// UncertaintyRouter_67061619360301
// MI455X (gfx1250) — compile-verified, matrix operand order checked by reading, not run on hardware
//
#include <hip/hip_runtime.h>
#include <hip/hip_bf16.h>
#include <math.h>

// ---------------------------------------------------------------------------
// Fused MoE uncertainty router for MI455X (gfx1250, wave32, WMMA).
//   logits = hs @ W_router^T + b_router          [tok, 64]
//   h1     = GELU(hs @ W_u1^T + b_u1)            [tok, 256]
//   u      = sigmoid(h1 @ W_u2^T + b_u2)         [tok]
//   k      = clamp(rint(1 + 3u), 1, 4)
//   top-4 of logits, zero-padded softmax per reference semantics.
// One block = 16 tokens, 20 waves; wave w owns combined columns [16w,16w+16).
// K-loop unrolled x2: two v_wmma_f32_16x16x32_bf16 per body, 16 b128 loads
// in flight for latency hiding.
// ---------------------------------------------------------------------------

typedef __attribute__((ext_vector_type(16))) __bf16 v16bf;
typedef __attribute__((ext_vector_type(8)))  float  v8f;

#define D_DIM   1024
#define E_DIM   64
#define DQ_DIM  256
#define NCOL    (E_DIM + DQ_DIM)   // 320 combined output columns
#define MTILE   16
#define KTILE   32
#define NWAVES  (NCOL / 16)        // 20
#define NTHREADS (NWAVES * 32)     // 640

__device__ __forceinline__ void pack8(v16bf& v, int base, float4 lo, float4 hi) {
    v[base + 0] = (__bf16)lo.x; v[base + 1] = (__bf16)lo.y;
    v[base + 2] = (__bf16)lo.z; v[base + 3] = (__bf16)lo.w;
    v[base + 4] = (__bf16)hi.x; v[base + 5] = (__bf16)hi.y;
    v[base + 6] = (__bf16)hi.z; v[base + 7] = (__bf16)hi.w;
}

__global__ __launch_bounds__(NTHREADS)
void UncertaintyRouter_kernel(const float* __restrict__ hs,
                              const float* __restrict__ Wr,
                              const float* __restrict__ br,
                              const float* __restrict__ Wu1,
                              const float* __restrict__ bu1,
                              const float* __restrict__ Wu2,
                              const float* __restrict__ bu2,
                              float* __restrict__ out,
                              int nTok)
{
    __shared__ float logits_s[MTILE][E_DIM + 4];   // padded row stride: 68 dwords
    __shared__ float u_acc[MTILE];

    const int tid  = threadIdx.x;
    const int wave = tid >> 5;
    const int lane = tid & 31;
    const int half = lane >> 4;     // wave32 half: selects which K chunks we hold
    const int l16  = lane & 15;
    const int t0   = blockIdx.x * MTILE;

    if (tid < MTILE) u_acc[tid] = 0.0f;
    __syncthreads();

    // This lane's combined output column and source weight row (B = W^T, W is
    // row-major [out, D], so a B fragment is a contiguous slice of a W row).
    const int  n        = wave * 16 + l16;
    const bool isRouter = (n < E_DIM);
    const float* wrow = isRouter ? (Wr  + (size_t)n           * D_DIM)
                                 : (Wu1 + (size_t)(n - E_DIM) * D_DIM);
    const float* arow = hs + (size_t)(t0 + l16) * D_DIM;

    // 16-bit operand layout: lanes 0-15 hold K {0..7, 16..23} of the 32-K tile,
    // lanes 16-31 hold K {8..15, 24..31}.  Two contiguous 8-float chunks.
    const int kc0 = half * 8;
    const int kc1 = kc0 + 16;

    v8f c = {};   // 16x16 f32 accumulator

    #pragma unroll 2
    for (int kb = 0; kb < D_DIM; kb += KTILE) {
        const float4* ap0 = (const float4*)(arow + kb + kc0);
        const float4* ap1 = (const float4*)(arow + kb + kc1);
        const float4* bp0 = (const float4*)(wrow + kb + kc0);
        const float4* bp1 = (const float4*)(wrow + kb + kc1);
        float4 a0 = ap0[0], a1 = ap0[1];
        float4 a2 = ap1[0], a3 = ap1[1];
        float4 b0 = bp0[0], b1 = bp0[1];
        float4 b2 = bp1[0], b3 = bp1[1];

        v16bf a, b;
        pack8(a, 0, a0, a1); pack8(a, 8, a2, a3);
        pack8(b, 0, b0, b1); pack8(b, 8, b2, b3);

        // D = A x B + C  (f32 accumulate)
        c = __builtin_amdgcn_wmma_f32_16x16x32_bf16(
                /*neg_a=*/false, a, /*neg_b=*/false, b,
                /*c_mod=*/(short)0, c, /*reuse_a=*/false, /*reuse_b=*/false);
    }

    // C layout: c[r] at lane L is element (M = r + 8*half, N = l16-of-tile).
    if (isRouter) {
        const float bias = br[n];
        #pragma unroll
        for (int r = 0; r < 8; ++r)
            logits_s[r + half * 8][n] = c[r] + bias;
    } else {
        const int   q    = n - E_DIM;
        const float bias = bu1[q];
        const float w2   = Wu2[q];
        #pragma unroll
        for (int r = 0; r < 8; ++r) {
            float x = c[r] + bias;
            // exact GELU: 0.5 x (1 + erf(x / sqrt(2)))
            float g = 0.5f * x * (1.0f + erff(x * 0.70710678118654752f));
            float p = g * w2;
            // reduce across the 16 lanes sharing this output row
            p += __shfl_xor(p, 1);
            p += __shfl_xor(p, 2);
            p += __shfl_xor(p, 4);
            p += __shfl_xor(p, 8);
            if (l16 == 0) atomicAdd(&u_acc[r + half * 8], p);
        }
    }
    __syncthreads();

    // ---- per-token epilogue: sigmoid -> k, top-4, zero-padded softmax ----
    if (tid < MTILE) {
        const int j = tid;
        const int t = t0 + j;

        float u  = 1.0f / (1.0f + expf(-(u_acc[j] + bu2[0])));
        float kf = 1.0f + 3.0f * u;
        int   kk = (int)rintf(kf);              // round-half-even, like jnp.round
        kk = kk < 1 ? 1 : (kk > 4 ? 4 : kk);

        float vals[4]; int idxs[4];
        unsigned long long taken = 0ull;
        #pragma unroll
        for (int s = 0; s < 4; ++s) {
            float best = -INFINITY; int bi = 0;
            for (int e = 0; e < E_DIM; ++e) {
                if ((taken >> e) & 1ull) continue;
                float v = logits_s[j][e];
                if (v > best) { best = v; bi = e; }
            }
            taken |= 1ull << bi;
            vals[s] = best; idxs[s] = bi;
        }

        float w[4]; float idf[4];
        #pragma unroll
        for (int s = 0; s < 4; ++s) {
            bool valid = (s < kk);
            w[s]   = valid ? vals[s]        : 0.0f;
            idf[s] = valid ? (float)idxs[s] : -1.0f;
        }
        float m  = fmaxf(fmaxf(w[0], w[1]), fmaxf(w[2], w[3]));
        float e0 = expf(w[0] - m), e1 = expf(w[1] - m);
        float e2 = expf(w[2] - m), e3 = expf(w[3] - m);
        float inv = 1.0f / (e0 + e1 + e2 + e3);

        float* outW = out;                         // [nTok,4] weights
        float* outI = out + (size_t)nTok * 4;      // [nTok,4] indices (as float)
        float* outK = out + (size_t)nTok * 8;      // [nTok]   k values (as float)

        outW[(size_t)t * 4 + 0] = e0 * inv;
        outW[(size_t)t * 4 + 1] = e1 * inv;
        outW[(size_t)t * 4 + 2] = e2 * inv;
        outW[(size_t)t * 4 + 3] = e3 * inv;
        outI[(size_t)t * 4 + 0] = idf[0];
        outI[(size_t)t * 4 + 1] = idf[1];
        outI[(size_t)t * 4 + 2] = idf[2];
        outI[(size_t)t * 4 + 3] = idf[3];
        outK[t] = (float)kk;
    }
}

extern "C" void kernel_launch(void* const* d_in, const int* in_sizes, int n_in,
                              void* d_out, int out_size, void* d_ws, size_t ws_size,
                              hipStream_t stream) {
    const float* hs  = (const float*)d_in[0];
    const float* Wr  = (const float*)d_in[1];
    const float* br  = (const float*)d_in[2];
    const float* Wu1 = (const float*)d_in[3];
    const float* bu1 = (const float*)d_in[4];
    const float* Wu2 = (const float*)d_in[5];
    const float* bu2 = (const float*)d_in[6];

    const int E  = in_sizes[2];          // 64
    const int Dd = in_sizes[1] / E;      // 1024
    const int nTok = in_sizes[0] / Dd;   // B*S = 32768

    const int blocks = nTok / MTILE;     // 2048
    UncertaintyRouter_kernel<<<blocks, NTHREADS, 0, stream>>>(
        hs, Wr, br, Wu1, bu1, Wu2, bu2, (float*)d_out, nTok);
}
